// FrequencyDeformableAttention_60189671686361
// MI455X (gfx1250) — matrix operand order repaired and checked by reading, not compiled, not run on hardware
//
#include <hip/hip_runtime.h>
#include <math.h>
#include <stdint.h>

#define B_   4
#define CIN  256
#define CO   32
#define HW   65536     // 256*256
#define H1   127
#define HP   41
#define SP   1681      // 41*41
#define CDIV(a,b) (((a)+(b)-1)/(b))

typedef __attribute__((ext_vector_type(16))) __bf16   v16bf;
typedef __attribute__((ext_vector_type(8)))  float    v8f;
typedef __attribute__((ext_vector_type(4)))  unsigned u32x4;
typedef __attribute__((ext_vector_type(8)))  unsigned u32x8;

static __device__ __forceinline__ __bf16 f2bf(float f) { return (__bf16)f; }
static __device__ __forceinline__ int clampi(int v, int lo, int hi) { return v < lo ? lo : (v > hi ? hi : v); }
static __device__ __forceinline__ unsigned rfl(unsigned v) { return (unsigned)__builtin_amdgcn_readfirstlane((int)v); }

// ---------------------------------------------------------------------------
// K1: conv1x1 256->32 as WMMA bf16 GEMM.  D(32 x 262144) = W(32x256) * X(256 x N)
// K-slab (32 x 128 f32, row stride 256KB) is DMA'd into LDS by the Tensor Data
// Mover (TDM) with pad_interval=128dw/pad_amount=2dw -> LDS pitch 130 floats
// (bank-conflict-free column reads). Wave 0 issues the D#, waits TENSORcnt.
// ---------------------------------------------------------------------------
__global__ __launch_bounds__(256) void k_conv1(const float* __restrict__ x,
                                               const float* __restrict__ w,
                                               const float* __restrict__ bias,
                                               float* __restrict__ c0) {
  __shared__ __bf16 wlds[CO * CIN];               // 16 KB weights bf16
  __shared__ __align__(16) float  xf[32 * 130];   // 16.6 KB TDM tile, pitch 130
  __shared__ __align__(32) __bf16 xbf[128 * 32];  // 8 KB transposed [n][k]
  const int tid = threadIdx.x;
  for (int i = tid; i < CO * CIN; i += 256) wlds[i] = f2bf(w[i]);

  const int q0 = blockIdx.x * 128;          // never crosses batch (512 blocks/batch)
  const int b  = q0 >> 16;
  const int p0 = q0 & 65535;
  const float* xb = x + (size_t)b * CIN * HW;

  const int lane = tid & 31, wave = tid >> 5;
  const int hi = (lane >> 4) & 1, l15 = lane & 15;

  v8f acc0 = {}; v8f acc1 = {};
  for (int k0 = 0; k0 < CIN; k0 += 32) {
    __syncthreads();
    if (wave == 0) {
      // --- Tensor DMA descriptor (D#) ---
      unsigned ldsoff = rfl((unsigned)(uintptr_t)&xf[0]);   // low 32b = LDS offset
      uint64_t ga = (uint64_t)(uintptr_t)(xb + (size_t)k0 * HW + p0);
      u32x4 g0;
      g0.x = 1u;                                  // count=1 (valid user descriptor)
      g0.y = ldsoff;                              // lds_addr
      g0.z = rfl((unsigned)ga);                   // global_addr[31:0]
      g0.w = (rfl((unsigned)(ga >> 32)) & 0x01FFFFFFu) | (2u << 30); // [56:32], type=2
      u32x8 g1;
      g1[0] = (2u << 16)    // data_size = 4 bytes
            | (1u << 20)    // pad_enable
            | (6u << 22)    // pad_interval: every 128 DWORDs
            | (1u << 25);   // pad_amount: 2 DWORDs  -> pitch 130 floats
      g1[1] = 0u;                       // barrier addr; tensor_dim0[15:0]=0
      g1[2] = 1u | (256u << 16);        // tensor_dim0[31:16]=1 (65536); tensor_dim1 lo16=256
      g1[3] = (128u << 16);             // tensor_dim1 hi16=0; tile_dim0=128
      g1[4] = 32u;                      // tile_dim1=32; tile_dim2=0
      g1[5] = 65536u;                   // tensor_dim0_stride[31:0]
      g1[6] = 0u;                       // stride hi16 / dim1_stride lo16
      g1[7] = 0u;
      asm volatile("tensor_load_to_lds %0, %1" :: "s"(g0), "s"(g1) : "memory");
      __builtin_amdgcn_s_wait_tensorcnt(0);
    }
    __syncthreads();
    // convert + transpose f32 tile -> bf16 [n][k] (conflict-free: pitch 130)
    for (int i = tid; i < 128 * 32; i += 256) {
      int n = i >> 5, k = i & 31;
      xbf[i] = f2bf(xf[k * 130 + n]);
    }
    __syncthreads();

    v16bf a0, a1;
#pragma unroll
    for (int e = 0; e < 16; ++e) {
      int koff = e + hi * 16;                              // contiguous K per lane half (matches B)
      a0[e] = wlds[l15 * CIN + k0 + koff];
      a1[e] = wlds[(16 + l15) * CIN + k0 + koff];
    }
    // B fragment: lane's 16 K-values contiguous -> 2x ds_load_b128
    v16bf bf = *(const v16bf*)(xbf + (wave * 16 + l15) * 32 + hi * 16);
    acc0 = __builtin_amdgcn_wmma_f32_16x16x32_bf16(false, a0, false, bf, (short)0, acc0, false, false);
    acc1 = __builtin_amdgcn_wmma_f32_16x16x32_bf16(false, a1, false, bf, (short)0, acc1, false, false);
  }

  const int n = p0 + wave * 16 + l15;
  float* outb = c0 + (size_t)b * CO * HW;
#pragma unroll
  for (int r = 0; r < 8; ++r) {
    int m0 = r + hi * 8;                                   // ISA C/D layout
    outb[(size_t)m0 * HW + n]        = acc0[r] + bias[m0];
    outb[(size_t)(16 + m0) * HW + n] = acc1[r] + bias[16 + m0];
  }
}

// ---------------------------------------------------------------------------
// K2: 3x3 stride-2 valid conv 32->32, 256x256 -> 127x127
// ---------------------------------------------------------------------------
__global__ void k_conv_s2(const float* __restrict__ c0, const float* __restrict__ w,
                          const float* __restrict__ bias, float* __restrict__ o) {
  int idx = blockIdx.x * 256 + threadIdx.x;
  const int total = B_ * CO * H1 * H1;
  if (idx >= total) return;
  int xx = idx % H1, t = idx / H1;
  int yy = t % H1;  t /= H1;
  int co = t % CO;
  int b  = t / CO;
  const float* inb = c0 + (size_t)b * CO * HW;
  float acc = bias[co];
  for (int ci = 0; ci < CO; ++ci) {
    const float* ip = inb + (size_t)ci * HW + (2 * yy) * 256 + 2 * xx;
    const float* wp = w + ((co * CO) + ci) * 9;
#pragma unroll
    for (int ki = 0; ki < 3; ++ki)
#pragma unroll
      for (int kj = 0; kj < 3; ++kj)
        acc += wp[ki * 3 + kj] * ip[ki * 256 + kj];
  }
  o[idx] = acc;
}

// ---------------------------------------------------------------------------
// K3: 7x7 stride-3 valid max pool, 127 -> 41
// ---------------------------------------------------------------------------
__global__ void k_pool(const float* __restrict__ in, float* __restrict__ o) {
  int idx = blockIdx.x * 256 + threadIdx.x;
  const int total = B_ * CO * SP;
  if (idx >= total) return;
  int px = idx % HP, t = idx / HP;
  int py = t % HP; t /= HP;
  const float* base = in + (size_t)t * H1 * H1;
  float m = -INFINITY;
#pragma unroll
  for (int i = 0; i < 7; ++i)
#pragma unroll
    for (int j = 0; j < 7; ++j)
      m = fmaxf(m, base[(py * 3 + i) * H1 + px * 3 + j]);
  o[idx] = m;
}

// ---------------------------------------------------------------------------
// Naive separable DFT (41 prime): forward row pass (along x)
// ---------------------------------------------------------------------------
__global__ void k_fft_row(const float* __restrict__ in, float* __restrict__ fre, float* __restrict__ fim) {
  int idx = blockIdx.x * 256 + threadIdx.x;
  if (idx >= B_ * CO * SP) return;
  int kx = idx % HP, t = idx / HP;
  int y = t % HP;  t /= HP;
  const float* base = in + (size_t)t * SP;
  float sr = 0.f, si = 0.f;
  const float wv = 6.2831853071795864f * (float)kx / 41.f;
  for (int xx = 0; xx < HP; ++xx) {
    float v = base[y * HP + xx];
    float sn, cs; __sincosf(wv * (float)xx, &sn, &cs);
    sr += v * cs; si -= v * sn;                // e^{-i theta}
  }
  fre[idx] = sr; fim[idx] = si;
}

// forward column pass (along y)
__global__ void k_fft_col(const float* __restrict__ fre, const float* __restrict__ fim,
                          float* __restrict__ gre, float* __restrict__ gim) {
  int idx = blockIdx.x * 256 + threadIdx.x;
  if (idx >= B_ * CO * SP) return;
  int kx = idx % HP, t = idx / HP;
  int ky = t % HP; t /= HP;
  const float* br = fre + (size_t)t * SP;
  const float* bi = fim + (size_t)t * SP;
  float sr = 0.f, si = 0.f;
  const float wv = 6.2831853071795864f * (float)ky / 41.f;
  for (int y = 0; y < HP; ++y) {
    float fr = br[y * HP + kx], fi = bi[y * HP + kx];
    float sn, cs; __sincosf(wv * (float)y, &sn, &cs);
    sr += fr * cs + fi * sn;                   // (fr+ifi)*e^{-i theta}
    si += fi * cs - fr * sn;
  }
  gre[idx] = sr; gim[idx] = si;
}

// mag + unit phase vector (cos(ph)=re/|.|, sin(ph)=im/|.|) -> avoids atan2
__global__ void k_magphase(const float* __restrict__ gre, const float* __restrict__ gim,
                           float* __restrict__ mag, float* __restrict__ cr, float* __restrict__ ci) {
  int idx = blockIdx.x * 256 + threadIdx.x;
  if (idx >= B_ * CO * SP) return;
  float re = gre[idx], im = gim[idx];
  float m = sqrtf(re * re + im * im);
  float inv = m > 0.f ? 1.f / m : 0.f;
  mag[idx] = m;
  cr[idx] = m > 0.f ? re * inv : 1.f;
  ci[idx] = im * inv;
}

// ---------------------------------------------------------------------------
// Offset conv: 3x3 pad1, 32->18 on 41x41
// ---------------------------------------------------------------------------
__global__ void k_off(const float* __restrict__ in, const float* __restrict__ w,
                      const float* __restrict__ bias, float* __restrict__ o) {
  int idx = blockIdx.x * 256 + threadIdx.x;
  if (idx >= B_ * 18 * SP) return;
  int xx = idx % HP, t = idx / HP;
  int yy = t % HP; t /= HP;
  int oc = t % 18;
  int b  = t / 18;
  const float* inb = in + (size_t)b * CO * SP;
  float acc = bias[oc];
  for (int ci = 0; ci < CO; ++ci) {
    const float* img = inb + (size_t)ci * SP;
    const float* wp = w + ((oc * CO) + ci) * 9;
#pragma unroll
    for (int ki = 0; ki < 3; ++ki) {
      int y2 = yy + ki - 1;
      if (y2 < 0 || y2 >= HP) continue;
#pragma unroll
      for (int kj = 0; kj < 3; ++kj) {
        int x2 = xx + kj - 1;
        if (x2 < 0 || x2 >= HP) continue;
        acc += wp[ki * 3 + kj] * img[y2 * HP + x2];
      }
    }
  }
  o[idx] = acc;
}

// ---------------------------------------------------------------------------
// Deformable 3x3 conv (32->32) + ReLU, bilinear sampling with zero padding
// ---------------------------------------------------------------------------
__global__ void k_deform(const float* __restrict__ in, const float* __restrict__ off,
                         const float* __restrict__ w, const float* __restrict__ bias,
                         float* __restrict__ o) {
  int idx = blockIdx.x * 256 + threadIdx.x;
  if (idx >= B_ * CO * SP) return;
  int xx = idx % HP, t = idx / HP;
  int yy = t % HP; t /= HP;
  int co = t % CO;
  int b  = t / CO;
  const float* inb  = in  + (size_t)b * CO * SP;
  const float* offb = off + (size_t)b * 18 * SP;
  float acc = bias[co];
#pragma unroll
  for (int k = 0; k < 9; ++k) {
    int ki = k / 3, kj = k % 3;
    float dy = offb[(size_t)(2 * k + 0) * SP + yy * HP + xx];
    float dx = offb[(size_t)(2 * k + 1) * SP + yy * HP + xx];
    float py = (float)yy + (float)(ki - 1) + dy;
    float px = (float)xx + (float)(kj - 1) + dx;
    float yf = floorf(py), xf = floorf(px);
    float wy = py - yf, wx = px - xf;
    int y0 = (int)yf, x0 = (int)xf;
    int y1 = y0 + 1, x1 = x0 + 1;
    float vy0 = (y0 >= 0 && y0 < HP) ? 1.f : 0.f;
    float vy1 = (y1 >= 0 && y1 < HP) ? 1.f : 0.f;
    float vx0 = (x0 >= 0 && x0 < HP) ? 1.f : 0.f;
    float vx1 = (x1 >= 0 && x1 < HP) ? 1.f : 0.f;
    int cy0 = clampi(y0, 0, HP - 1), cy1 = clampi(y1, 0, HP - 1);
    int cx0 = clampi(x0, 0, HP - 1), cx1 = clampi(x1, 0, HP - 1);
    float w00 = (1.f - wy) * (1.f - wx) * vy0 * vx0;
    float w01 = (1.f - wy) * wx * vy0 * vx1;
    float w10 = wy * (1.f - wx) * vy1 * vx0;
    float w11 = wy * wx * vy1 * vx1;
    for (int ci = 0; ci < CO; ++ci) {
      const float* img = inb + (size_t)ci * SP;
      float s = w00 * img[cy0 * HP + cx0] + w01 * img[cy0 * HP + cx1] +
                w10 * img[cy1 * HP + cx0] + w11 * img[cy1 * HP + cx1];
      acc += w[((co * CO) + ci) * 9 + k] * s;
    }
  }
  o[idx] = fmaxf(acc, 0.f);
}

// ---------------------------------------------------------------------------
// Inverse DFT column pass: spec = a*(cr + i*ci), sum over ky with e^{+i theta}
// ---------------------------------------------------------------------------
__global__ void k_ifft_col(const float* __restrict__ a, const float* __restrict__ cr,
                           const float* __restrict__ ci, float* __restrict__ fre,
                           float* __restrict__ fim) {
  int idx = blockIdx.x * 256 + threadIdx.x;
  if (idx >= B_ * CO * SP) return;
  int kx = idx % HP, t = idx / HP;
  int y = t % HP; t /= HP;
  const size_t base = (size_t)t * SP;
  float sr = 0.f, si = 0.f;
  const float wv = 6.2831853071795864f * (float)y / 41.f;
  for (int ky = 0; ky < HP; ++ky) {
    float av = a[base + ky * HP + kx];
    float ar = av * cr[base + ky * HP + kx];
    float ai = av * ci[base + ky * HP + kx];
    float sn, cs; __sincosf(wv * (float)ky, &sn, &cs);
    sr += ar * cs - ai * sn;                  // (ar+iai)*e^{+i theta}
    si += ar * sn + ai * cs;
  }
  fre[idx] = sr; fim[idx] = si;
}

// inverse row pass, real part, 1/(41*41) normalization
__global__ void k_ifft_row(const float* __restrict__ fre, const float* __restrict__ fim,
                           float* __restrict__ rec) {
  int idx = blockIdx.x * 256 + threadIdx.x;
  if (idx >= B_ * CO * SP) return;
  int xx = idx % HP, t = idx / HP;
  int y = t % HP; t /= HP;
  const size_t base = (size_t)t * SP;
  float s = 0.f;
  const float wv = 6.2831853071795864f * (float)xx / 41.f;
  for (int kx = 0; kx < HP; ++kx) {
    float sn, cs; __sincosf(wv * (float)kx, &sn, &cs);
    s += fre[base + y * HP + kx] * cs - fim[base + y * HP + kx] * sn;
  }
  rec[idx] = s * (1.f / 1681.f);
}

// ---------------------------------------------------------------------------
// Bilinear resize 41x41 -> 256x256 (half-pixel centers, clamped edges)
// ---------------------------------------------------------------------------
__global__ void k_resize(const float* __restrict__ rec, float* __restrict__ up) {
  int idx = blockIdx.x * 256 + threadIdx.x;
  if (idx >= B_ * CO * HW) return;
  int xx = idx & 255;
  int yy = (idx >> 8) & 255;
  int t  = idx >> 16;  // b*32+c
  const float* img = rec + (size_t)t * SP;
  const float scale = 41.f / 256.f;
  float sy = ((float)yy + 0.5f) * scale - 0.5f;
  float sx = ((float)xx + 0.5f) * scale - 0.5f;
  float yf = floorf(sy), xf = floorf(sx);
  float wy = sy - yf, wx = sx - xf;
  int y0 = clampi((int)yf, 0, HP - 1), y1 = clampi((int)yf + 1, 0, HP - 1);
  int x0 = clampi((int)xf, 0, HP - 1), x1 = clampi((int)xf + 1, 0, HP - 1);
  float v = (1.f - wy) * ((1.f - wx) * img[y0 * HP + x0] + wx * img[y0 * HP + x1]) +
            wy * ((1.f - wx) * img[y1 * HP + x0] + wx * img[y1 * HP + x1]);
  up[idx] = v;
}

// ---------------------------------------------------------------------------
// branch 1x1 (32->32) on c0, added in-place into up:  up += branch(c0)
// ---------------------------------------------------------------------------
__global__ void k_branch_add(const float* __restrict__ c0, const float* __restrict__ bw,
                             const float* __restrict__ bb, float* __restrict__ up) {
  int idx = blockIdx.x * 256 + threadIdx.x;
  if (idx >= B_ * CO * HW) return;
  int p = idx & 65535;
  int c = (idx >> 16) & 31;
  int b = idx >> 21;
  const float* c0b = c0 + (size_t)b * CO * HW;
  float acc = bb[c];
  for (int j = 0; j < CO; ++j) acc += bw[c * CO + j] * c0b[(size_t)j * HW + p];
  up[idx] += acc;
}

// ---------------------------------------------------------------------------
// K-final: conv2 1x1 (32->256) as WMMA bf16 GEMM, fused sigmoid gate:
//   out = x * sigmoid(W2 * t + b2).  Block: 8 waves cover all M=256, N=16.
// ---------------------------------------------------------------------------
__global__ __launch_bounds__(256) void k_conv2_fused(const float* __restrict__ t,
                                                     const float* __restrict__ w,
                                                     const float* __restrict__ bias,
                                                     const float* __restrict__ x,
                                                     float* __restrict__ out) {
  __shared__ __bf16 wlds[CIN * CO];                 // 16 KB: 256x32 weights
  __shared__ __align__(32) __bf16 tbf[16 * 32];     // [n][k] transposed tile
  const int tid = threadIdx.x;
  for (int i = tid; i < CIN * CO; i += 256) wlds[i] = f2bf(w[i]);

  const int q0 = blockIdx.x * 16;
  const int b  = q0 >> 16;
  const int p0 = q0 & 65535;
  const float* tb = t + (size_t)b * CO * HW;
  for (int i = tid; i < 32 * 16; i += 256) {
    int kk = i >> 4, nn = i & 15;
    tbf[nn * 32 + kk] = f2bf(tb[(size_t)kk * HW + p0 + nn]);
  }
  const int lane = tid & 31, wave = tid >> 5;
  const int hi = (lane >> 4) & 1, l15 = lane & 15;
  // prefetch the gate operand x for this block's rows -> global_prefetch_b8
  __builtin_prefetch(x + (size_t)b * CIN * HW + (size_t)(wave * 32) * HW + p0, 0, 1);
  __syncthreads();

  v16bf a0, a1;
#pragma unroll
  for (int e = 0; e < 16; ++e) {
    int koff = e + hi * 16;
    a0[e] = wlds[(wave * 32 + l15) * CO + koff];
    a1[e] = wlds[(wave * 32 + 16 + l15) * CO + koff];
  }
  v16bf bfr = *(const v16bf*)(tbf + l15 * 32 + hi * 16);
  v8f acc0 = {}, acc1 = {};
  acc0 = __builtin_amdgcn_wmma_f32_16x16x32_bf16(false, a0, false, bfr, (short)0, acc0, false, false);
  acc1 = __builtin_amdgcn_wmma_f32_16x16x32_bf16(false, a1, false, bfr, (short)0, acc1, false, false);

  const float* xb = x + (size_t)b * CIN * HW;
  float* ob = out + (size_t)b * CIN * HW;
  const int n = p0 + l15;
#pragma unroll
  for (int r = 0; r < 8; ++r) {
    int m  = wave * 32 + r + hi * 8;
    int m2 = m + 16;
    float v1 = acc0[r] + bias[m];
    float v2 = acc1[r] + bias[m2];
    ob[(size_t)m  * HW + n] = xb[(size_t)m  * HW + n] * (1.f / (1.f + __expf(-v1)));
    ob[(size_t)m2 * HW + n] = xb[(size_t)m2 * HW + n] * (1.f / (1.f + __expf(-v2)));
  }
}

// ---------------------------------------------------------------------------
extern "C" void kernel_launch(void* const* d_in, const int* in_sizes, int n_in,
                              void* d_out, int out_size, void* d_ws, size_t ws_size,
                              hipStream_t stream) {
  const float* x        = (const float*)d_in[0];
  const float* conv1_w  = (const float*)d_in[1];
  const float* conv1_b  = (const float*)d_in[2];
  const float* stride_w = (const float*)d_in[3];
  const float* stride_b = (const float*)d_in[4];
  const float* branch_w = (const float*)d_in[5];
  const float* branch_b = (const float*)d_in[6];
  const float* off_w1   = (const float*)d_in[7];
  const float* off_b1   = (const float*)d_in[8];
  const float* dcn_w1   = (const float*)d_in[9];
  const float* dcn_b1   = (const float*)d_in[10];
  const float* off_w2   = (const float*)d_in[11];
  const float* off_b2   = (const float*)d_in[12];
  const float* dcn_w2   = (const float*)d_in[13];
  const float* dcn_b2   = (const float*)d_in[14];
  const float* off_w3   = (const float*)d_in[15];
  const float* off_b3   = (const float*)d_in[16];
  const float* dcn_w3   = (const float*)d_in[17];
  const float* dcn_b3   = (const float*)d_in[18];
  const float* conv2_w  = (const float*)d_in[19];
  const float* conv2_b  = (const float*)d_in[20];
  float* out = (float*)d_out;

  // workspace layout (floats)
  const size_t S_IMG = (size_t)B_ * CO * HW;      // 8,388,608
  const size_t S_T1  = (size_t)B_ * CO * H1 * H1; // 2,064,512
  const size_t S_P   = (size_t)B_ * CO * SP;      // 215,168
  const size_t S_OFF = (size_t)B_ * 18 * SP;      // 121,032
  float* ws  = (float*)d_ws;
  float* c0   = ws;
  float* t1   = c0  + S_IMG;
  float* pool = t1  + S_T1;
  float* fre  = pool + S_P;
  float* fim  = fre + S_P;
  float* gre  = fim + S_P;
  float* gim  = gre + S_P;
  float* mag  = gim + S_P;
  float* crb  = mag + S_P;
  float* cib  = crb + S_P;
  float* a0b  = cib + S_P;
  float* a1b  = a0b + S_P;
  float* offb = a1b + S_P;
  float* rec  = offb + S_OFF;
  float* up   = rec + S_P;     // also serves as 't' after in-place branch add

  const int gP   = (int)CDIV(S_P, 256);
  const int gOFF = (int)CDIV(S_OFF, 256);
  const int gIMG = (int)CDIV(S_IMG, 256);

  // 1) big GEMM: c0 = conv1x1(x)   [TDM -> LDS -> WMMA bf16]
  k_conv1<<<2048, 256, 0, stream>>>(x, conv1_w, conv1_b, c0);
  // 2) 3x3 stride-2 conv -> 127x127
  k_conv_s2<<<(int)CDIV(S_T1, 256), 256, 0, stream>>>(c0, stride_w, stride_b, t1);
  // 3) 7x7/3 max pool -> 41x41
  k_pool<<<gP, 256, 0, stream>>>(t1, pool);
  // 4) forward DFT (separable naive; 41 prime)
  k_fft_row<<<gP, 256, 0, stream>>>(pool, fre, fim);
  k_fft_col<<<gP, 256, 0, stream>>>(fre, fim, gre, gim);
  k_magphase<<<gP, 256, 0, stream>>>(gre, gim, mag, crb, cib);
  // 5) deformable conv chain on magnitude
  k_off<<<gOFF, 256, 0, stream>>>(mag, off_w1, off_b1, offb);
  k_deform<<<gP, 256, 0, stream>>>(mag, offb, dcn_w1, dcn_b1, a0b);
  k_off<<<gOFF, 256, 0, stream>>>(a0b, off_w2, off_b2, offb);
  k_deform<<<gP, 256, 0, stream>>>(a0b, offb, dcn_w2, dcn_b2, a1b);
  k_off<<<gOFF, 256, 0, stream>>>(a1b, off_w3, off_b3, offb);
  k_deform<<<gP, 256, 0, stream>>>(a1b, offb, dcn_w3, dcn_b3, a0b);
  // 6) inverse DFT with preserved phase
  k_ifft_col<<<gP, 256, 0, stream>>>(a0b, crb, cib, fre, fim);
  k_ifft_row<<<gP, 256, 0, stream>>>(fre, fim, rec);
  // 7) bilinear upsample 41 -> 256
  k_resize<<<gIMG, 256, 0, stream>>>(rec, up);
  // 8) t = branch(c0) + up  (in place into up)
  k_branch_add<<<gIMG, 256, 0, stream>>>(c0, branch_w, branch_b, up);
  // 9) big GEMM + fused gate: out = x * sigmoid(conv2(t))   [WMMA bf16]
  k_conv2_fused<<<16384, 256, 0, stream>>>(up, conv2_w, conv2_b, x, out);
}
